// RNN_3169685865233
// MI455X (gfx1250) — matrix operand order repaired and checked by reading, not compiled, not run on hardware
//
#include <hip/hip_runtime.h>
#include <hip/hip_bf16.h>
#include <math.h>

// ---------------------------------------------------------------------------
// Vanilla tanh-RNN, T=4096 B=64 I=H=256, fp32 in/out, bf16 WMMA internally.
//
// Kernel 1 (rnn_xproj):  xp = x @ W_ih^T + (b_ih + b_hh)      -> written to d_out
// Kernel 2 (rnn_recur):  for t: h = tanh(xp_t + h @ W_hh^T)   -> overwrites d_out[t]
// ---------------------------------------------------------------------------

typedef __attribute__((ext_vector_type(16))) __bf16 v16bf;
typedef __attribute__((ext_vector_type(8)))  __bf16 v8bf;
typedef __attribute__((ext_vector_type(4)))  __bf16 v4bf;
typedef __attribute__((ext_vector_type(8)))  float  v8f;
typedef __attribute__((ext_vector_type(4)))  int    v4i;

#define TT 4096
#define BB 64
#define HH 256
#define KTILES 8          // 256 / 32
#define NT 16             // 256 / 16  (N tiles)
#define A_STRIDE 264      // padded bf16 row stride (banks shift 4/row)
#define XSTR 260          // padded fp32 row stride for the xp LDS ring

// ---------------------------------------------------------------------------
// CDNA5 async global->LDS path (ASYNCcnt-tracked); sync fallback keeps the
// file compiling on toolchains without the builtin.
// ---------------------------------------------------------------------------
#if __has_builtin(__builtin_amdgcn_global_load_async_to_lds_b128)
#define HAS_ASYNC_LDS 1
#else
#define HAS_ASYNC_LDS 0
#endif

__device__ __forceinline__ void xp_copy16(const float* g, float* l) {
#if HAS_ASYNC_LDS
    __builtin_amdgcn_global_load_async_to_lds_b128(
        (__attribute__((address_space(1))) v4i*)g,
        (__attribute__((address_space(3))) v4i*)l, 0, 0);
#else
    *(float4*)l = *(const float4*)g;        // synchronous fallback
#endif
}

__device__ __forceinline__ void xp_wait4() {
#if HAS_ASYNC_LDS
#if __has_builtin(__builtin_amdgcn_s_wait_asynccnt)
    __builtin_amdgcn_s_wait_asynccnt(4);    // step-t loads done, t+1 in flight
#else
    asm volatile("s_wait_asynccnt 0x4" ::: "memory");
#endif
#endif
}

__device__ __forceinline__ void lds_drain() {
#if HAS_ASYNC_LDS
    // Retire our DS reads of the ring buffer before the async refill can land,
    // and pin issue order (memory clobber) so the loads are not sunk below.
    asm volatile("s_wait_dscnt 0x0" ::: "memory");
#endif
}

// gfx1250 has a hardware tanh (TRANS op, co-executes with VALU) — big win on
// the 4096-step serial chain vs the ocml polynomial.
__device__ __forceinline__ float fast_tanh(float x) {
#if __has_builtin(__builtin_amdgcn_tanhf)
    return __builtin_amdgcn_tanhf(x);
#else
    return tanhf(x);
#endif
}

// Packed B-operand layout used here for v_wmma_f32_16x16x32_bf16 (contiguous k, same as A):
//   lanes 0-15 hold K=0..15 (8 VGPRs x 2 halves), lanes 16-31 hold K=16..31,
//   N = lane & 15.  Each fragment = 16 contiguous halfs per lane.
__device__ __forceinline__ int bpack_idx(int n, int k) {
    int tn = n >> 4, kk = k >> 5;
    int lane = (n & 15) + (((k & 31) >= 16) ? 16 : 0);
    return (((kk * NT + tn) * 32) + lane) * 16 + (k & 15);
}

__device__ __forceinline__ v16bf cat16(v8bf lo, v8bf hi) {
    return __builtin_shufflevector(lo, hi, 0, 1, 2, 3, 4, 5, 6, 7,
                                           8, 9, 10, 11, 12, 13, 14, 15);
}

__device__ __forceinline__ v16bf load_frag(const __bf16* p) {
    return cat16(*(const v8bf*)p, *(const v8bf*)(p + 8));
}

// ---------------------------------------------------------------------------
// Phase 1: xp[r,n] = sum_k x[r,k] * W_ih[n,k] + bias[n]
// 1024 threads (32 waves); tile M=128, N=256, K=256; wave = 1 mTile x 4 nTiles
// (acc = 32 VGPRs -> the 4-fragment B batch stays register-resident).
// HBM-bound: streams x once; W_ih stays L2-resident (192 MB L2).
// ---------------------------------------------------------------------------
__global__ __launch_bounds__(1024) void rnn_xproj(
        const float* __restrict__ x,
        const float* __restrict__ W_ih,
        const float* __restrict__ b_ih,
        const float* __restrict__ b_hh,
        float* __restrict__ out)
{
    extern __shared__ char smem[];
    __bf16* As   = (__bf16*)smem;                    // [128][A_STRIDE] bf16
    __bf16* Bp   = As + 128 * A_STRIDE;              // packed W_ih
    float*  bias = (float*)(Bp + NT * KTILES * 32 * 16);

    const int tid  = threadIdx.x;
    const int row0 = blockIdx.x * 128;

    if (tid < HH) bias[tid] = b_ih[tid] + b_hh[tid];

    // Stage x tile -> bf16 LDS (float4 vectorized, fully coalesced HBM stream)
    const float4* xv = (const float4*)(x + (size_t)row0 * HH);
    #pragma unroll
    for (int it = 0; it < 8; ++it) {
        int i4 = tid + it * 1024;                    // 0 .. 8191
        float4 v = xv[i4];
        int r = i4 >> 6, c4 = i4 & 63;
        v4bf b;
        b[0] = (__bf16)v.x; b[1] = (__bf16)v.y;
        b[2] = (__bf16)v.z; b[3] = (__bf16)v.w;
        *(v4bf*)(As + r * A_STRIDE + c4 * 4) = b;
    }

    // Pack W_ih into WMMA B layout: B[k][n] = W_ih[n][k]
    for (int e = tid; e < HH * HH; e += 1024) {
        int n = e >> 8, k = e & 255;
        Bp[bpack_idx(n, k)] = (__bf16)W_ih[e];
    }
    __syncthreads();

    const int wave   = tid >> 5;
    const int lane   = tid & 31;
    const int mT     = wave >> 2;                    // 0..7
    const int nBase  = (wave & 3) * 4;               // 4 nTiles per wave
    const int laneLo = lane & 15;
    const int aRow   = mT * 16 + laneLo;
    const int aK     = (lane < 16) ? 0 : 16;

    // acc initialized with the bias (saves the epilogue adds)
    v8f acc[4];
    #pragma unroll
    for (int i = 0; i < 4; ++i) {
        float bv = bias[(nBase + i) * 16 + laneLo];
        #pragma unroll
        for (int r = 0; r < 8; ++r) acc[i][r] = bv;
    }

    #pragma unroll
    for (int kk = 0; kk < KTILES; ++kk) {
        v16bf Af = load_frag(As + aRow * A_STRIDE + kk * 32 + aK);
        v16bf Bf[4];
        #pragma unroll
        for (int i = 0; i < 4; ++i)
            Bf[i] = load_frag(Bp + (((kk * NT + (nBase + i)) * 32) + lane) * 16);
        #pragma unroll
        for (int i = 0; i < 4; ++i)
            acc[i] = __builtin_amdgcn_wmma_f32_16x16x32_bf16(
                false, Af, false, Bf[i], (short)0, acc[i], false, false);
    }

    // C layout: m = r + (lane>=16)*8, n = tile*16 + (lane&15)
    const int mBase = mT * 16 + ((lane < 16) ? 0 : 8);
    #pragma unroll
    for (int i = 0; i < 4; ++i) {
        int n = (nBase + i) * 16 + laneLo;
        #pragma unroll
        for (int r = 0; r < 8; ++r)
            out[(size_t)(row0 + mBase + r) * HH + n] = acc[i][r];
    }
}

// ---------------------------------------------------------------------------
// Phase 2: sequential scan, ONE workgroup (1024 threads, 32 waves) owning the
// state in LDS: packed W_hh (128 KB) + h bf16 (33 KB) + xp ring (2 x 65 KB).
// Wave w: mTile = w&3, tiles tn = (w>>2)*2 + {0,1}.  16 WMMA / wave / step.
// xp[t+2] is async-copied global->LDS (per-wave slice -> per-wave ASYNCcnt
// waits, no extra barriers).  B fragments for kk=0,1 are held in registers
// across the whole loop so the post-barrier dependence is just the A load.
// ---------------------------------------------------------------------------
__global__ __launch_bounds__(1024) void rnn_recur(
        const float* __restrict__ W_hh,
        float* __restrict__ out)      // holds xp on entry, h on exit
{
    extern __shared__ char smem[];
    __bf16* Wp = (__bf16*)smem;                      // 65536 halfs (128 KB)
    __bf16* Hs = Wp + NT * KTILES * 32 * 16;         // [64][A_STRIDE] bf16
    float*  X0 = (float*)(smem + (NT * KTILES * 32 * 16 + BB * A_STRIDE) * 2);
    float*  X1 = X0 + BB * XSTR;                     // xp ring, 2 buffers

    const int tid = threadIdx.x;

    // Pack W_hh into WMMA B layout: B[k][n] = W_hh[n][k]
    for (int e = tid; e < HH * HH; e += 1024) {
        int n = e >> 8, k = e & 255;
        Wp[bpack_idx(n, k)] = (__bf16)W_hh[e];
    }
    for (int e = tid; e < BB * A_STRIDE; e += 1024) Hs[e] = (__bf16)0.0f;
    __syncthreads();

    const int wave   = tid >> 5;
    const int lane   = tid & 31;
    const int m0     = (wave & 3) * 16;
    const int np     = wave >> 2;                    // 0..7
    const int tn0    = np * 2, tn1 = np * 2 + 1;
    const int laneLo = lane & 15;
    const int aK     = (lane < 16) ? 0 : 16;
    const int mBase  = m0 + ((lane < 16) ? 0 : 8);
    const int n0     = tn0 * 16 + laneLo;
    const int n1     = tn1 * 16 + laneLo;

    const __bf16* hrow = Hs + (m0 + laneLo) * A_STRIDE + aK;

    // Loop-invariant B fragments for kk=0,1 pinned in registers.
    v16bf PB0[2], PB1[2];
    #pragma unroll
    for (int kk = 0; kk < 2; ++kk) {
        PB0[kk] = load_frag(Wp + ((kk * NT + tn0) * 32 + lane) * 16);
        PB1[kk] = load_frag(Wp + ((kk * NT + tn1) * 32 + lane) * 16);
    }

    // This wave's xp slice: rows m0..m0+15, cols np*32..np*32+31  (2 KB).
    // Per lane: one 64-byte run -> 4 async b128 copies per step.
    const int srow = m0 + (lane >> 1);
    const int scol = np * 32 + (lane & 1) * 16;

    // Prime the ring: xp[0] -> X0, xp[1] -> X1
    #pragma unroll
    for (int j = 0; j < 4; ++j)
        xp_copy16(out + (size_t)srow * HH + scol + j * 4,
                  X0 + srow * XSTR + scol + j * 4);
    #pragma unroll
    for (int j = 0; j < 4; ++j)
        xp_copy16(out + (size_t)(BB * HH) + (size_t)srow * HH + scol + j * 4,
                  X1 + srow * XSTR + scol + j * 4);

    for (int t = 0; t < TT; ++t) {
        float* Xc = (t & 1) ? X1 : X0;

        // xp[t] has landed (in-order async completion; t+1 stays in flight)
        xp_wait4();

        // acc starts at xp (C operand of the first WMMA does the add)
        v8f acc0, acc1;
        #pragma unroll
        for (int r = 0; r < 8; ++r) {
            acc0[r] = Xc[(mBase + r) * XSTR + n0];
            acc1[r] = Xc[(mBase + r) * XSTR + n1];
        }

        // Refill this buffer with xp[t+2] once our reads have retired
        lds_drain();
        {
            int tp = (t + 2 < TT) ? (t + 2) : (TT - 1);
            const float* g = out + (size_t)tp * BB * HH + (size_t)srow * HH + scol;
            float* l = Xc + srow * XSTR + scol;
            #pragma unroll
            for (int j = 0; j < 4; ++j) xp_copy16(g + j * 4, l + j * 4);
        }

        // GEMM: kk=0,1 use pinned B; kk>=2 double-buffered under WMMA cover
        v16bf Af[2], Bt0[2], Bt1[2];
        Af[0] = load_frag(hrow);
        #pragma unroll
        for (int kk = 0; kk < KTILES; ++kk) {
            const int cur = kk & 1, nxt = cur ^ 1;
            if (kk < KTILES - 1) {
                Af[nxt] = load_frag(hrow + (kk + 1) * 32);
                if (kk + 1 >= 2) {
                    Bt0[nxt] = load_frag(Wp + (((kk + 1) * NT + tn0) * 32 + lane) * 16);
                    Bt1[nxt] = load_frag(Wp + (((kk + 1) * NT + tn1) * 32 + lane) * 16);
                }
            }
            v16bf b0 = (kk < 2) ? PB0[kk] : Bt0[cur];
            v16bf b1 = (kk < 2) ? PB1[kk] : Bt1[cur];
            acc0 = __builtin_amdgcn_wmma_f32_16x16x32_bf16(
                false, Af[cur], false, b0, (short)0, acc0, false, false);
            acc1 = __builtin_amdgcn_wmma_f32_16x16x32_bf16(
                false, Af[cur], false, b1, (short)0, acc1, false, false);
        }

        // h = tanh(acc); fp32 -> out[t], bf16 -> LDS for next step's A operand
        float* po = out + (size_t)t * BB * HH;
        #pragma unroll
        for (int r = 0; r < 8; ++r) {
            float h0 = fast_tanh(acc0[r]);
            float h1 = fast_tanh(acc1[r]);
            po[(size_t)(mBase + r) * HH + n0] = h0;
            po[(size_t)(mBase + r) * HH + n1] = h1;
            Hs[(mBase + r) * A_STRIDE + n0] = (__bf16)h0;
            Hs[(mBase + r) * A_STRIDE + n1] = (__bf16)h1;
        }

        __syncthreads();   // h fully in LDS before next step's A-fragment loads
    }
}

// ---------------------------------------------------------------------------
extern "C" void kernel_launch(void* const* d_in, const int* in_sizes, int n_in,
                              void* d_out, int out_size, void* d_ws, size_t ws_size,
                              hipStream_t stream)
{
    const float* x    = (const float*)d_in[0];
    const float* W_ih = (const float*)d_in[1];
    const float* W_hh = (const float*)d_in[2];
    const float* b_ih = (const float*)d_in[3];
    const float* b_hh = (const float*)d_in[4];
    float* out = (float*)d_out;

    const size_t packBytes = (size_t)NT * KTILES * 32 * 16 * sizeof(__bf16); // 128 KB
    const size_t shm1 = (size_t)128 * A_STRIDE * sizeof(__bf16) + packBytes
                        + HH * sizeof(float);
    const size_t shm2 = packBytes + (size_t)BB * A_STRIDE * sizeof(__bf16)
                        + (size_t)2 * BB * XSTR * sizeof(float);   // ~291 KB

    // Phase 1: (T*B)/128 = 2048 workgroups, streams x once from HBM.
    rnn_xproj<<<dim3((TT * BB) / 128), dim3(1024), shm1, stream>>>(
        x, W_ih, b_ih, b_hh, out);

    // Phase 2: single WGP owns the recurrence.
    rnn_recur<<<dim3(1), dim3(1024), shm2, stream>>>(W_hh, out);
}
